// ConLoss_18245021073784
// MI455X (gfx1250) — matrix operand order repaired and checked by reading, not compiled, not run on hardware
//
#include <hip/hip_runtime.h>
#include <hip/hip_fp16.h>

typedef __attribute__((ext_vector_type(16))) _Float16 v16h;
typedef __attribute__((ext_vector_type(8)))  float    v8f;
typedef __attribute__((ext_vector_type(4)))  unsigned int v4u;
typedef __attribute__((ext_vector_type(8)))  int      v8i;
typedef __attribute__((ext_vector_type(4)))  int      v4i;

#define N_ROWS   8192
#define DIM      128
#define JSPLIT   8
#define LDS_H    136       // 128 halves + 8-half pad (TDM pad: 4 DWORDs per 64 DWORDs)
#define NPART    (64 * JSPLIT)

#define K_EXP2   2.885390081777926f   // 2 * log2(e)  : exp(2c) = exp2(c*K_EXP2)
#define LN2      0.693147180559945f

// ---------------------------------------------------------------------------
// TDM: DMA one 16x128 f16 tile (row stride DIM) into LDS with a 16B pad every
// 256B so LDS rows land at stride LDS_H halves. Issued by one wave; tracked
// with TENSORcnt.  Descriptor layout per CDNA5 ISA 8.3-8.6.
// ---------------------------------------------------------------------------
__device__ __forceinline__ void tdm_load_tile(const _Float16* gsrc, unsigned ldsOff) {
    unsigned long long ga = (unsigned long long)gsrc;
    v4u g0 = { 1u,                                   // count=1, user desc
               ldsOff,                               // lds_addr (bytes)
               (unsigned)ga,                         // global_addr[31:0]
               (unsigned)((ga >> 32) & 0x1FFFFFFu) | 0x80000000u };  // addr[56:32] | type=2
    v8i g1 = { (int)((1u << 16)      // data_size = 2 bytes
                   | (1u << 20)      // pad_enable
                   | (5u << 22)      // pad_interval: 64 DWORDs (256B)
                   | (3u << 25)),    // pad_amount : 4 DWORDs (16B)
               (int)(128u << 16),    // tensor_dim0 = 128 (low16 in [63:48])
               (int)(8192u << 16),   // tensor_dim1 = 8192 (low16 in [95:80])
               (int)(128u << 16),    // tile_dim0 = 128
               16,                   // tile_dim1 = 16
               128,                  // tensor_dim0_stride = 128
               0, 0 };
    v4i g2 = { 0, 0, 0, 0 };
    v4i g3 = { 0, 0, 0, 0 };
#if defined(__clang_major__) && __clang_major__ >= 23
    v8i g4 = { 0, 0, 0, 0, 0, 0, 0, 0 };
    __builtin_amdgcn_tensor_load_to_lds(g0, g1, g2, g3, g4, 0);
#else
    __builtin_amdgcn_tensor_load_to_lds(g0, g1, g2, g3, 0);
#endif
}

// ---------------------------------------------------------------------------
// Kernel 1: row-normalize fp32 features -> f16, one wave (32 lanes) per row.
// ---------------------------------------------------------------------------
__global__ __launch_bounds__(256)
void conloss_norm_kernel(const float* __restrict__ F, _Float16* __restrict__ out) {
    const int wave = threadIdx.x >> 5;
    const int lane = threadIdx.x & 31;
    const int row  = blockIdx.x * 8 + wave;

    float4 v = ((const float4*)(F + (size_t)row * DIM))[lane];
    float s = v.x * v.x + v.y * v.y + v.z * v.z + v.w * v.w;
#pragma unroll
    for (int m = 16; m >= 1; m >>= 1) s += __shfl_xor(s, m, 32);
    const float scale = 1.0f / fmaxf(sqrtf(s), 1e-8f);

    union { _Float16 h[4]; uint2 u; } pk;
    pk.h[0] = (_Float16)(v.x * scale);
    pk.h[1] = (_Float16)(v.y * scale);
    pk.h[2] = (_Float16)(v.z * scale);
    pk.h[3] = (_Float16)(v.w * scale);
    *(uint2*)(out + (size_t)row * DIM + lane * 4) = pk.u;
}

// ---------------------------------------------------------------------------
// Kernel 2: fused WMMA GEMM + loss reductions, TDM-staged B tiles.
//   pass==1: Epart[by][i] = sum_{j in chunk} (pos ? 1 : exp(sim_ij))
//   pass==2: per-block partials of sum_pos [log(exp(sim)+E_i) - sim] and #pos
// ---------------------------------------------------------------------------
__global__ __launch_bounds__(256)
void conloss_sim_kernel(const _Float16* __restrict__ fnh,
                        const int*      __restrict__ labels,
                        float*          __restrict__ Epart,
                        const float*    __restrict__ E,
                        float*          __restrict__ lossPart,
                        float*          __restrict__ npPart,
                        int pass) {
    __shared__ _Float16 ldsB[2][16 * LDS_H];
    __shared__ float redL[8], redN[8];

    const int wave   = threadIdx.x >> 5;
    const int lane   = threadIdx.x & 31;
    const int laneHi = lane >> 4;        // 0: M rows 0..7 | 1: M rows 8..15
    const int laneLo = lane & 15;        // N column within tile
    const int i0     = blockIdx.x * 128 + wave * 16;
    const int jBase  = blockIdx.y * (N_ROWS / JSPLIT);
    const int nTiles = (N_ROWS / JSPLIT) / 16;

    const unsigned ldsOff0 = (unsigned)(unsigned long long)&ldsB[0][0];
    const unsigned ldsOff1 = (unsigned)(unsigned long long)&ldsB[1][0];

    // --- A fragments: rows i0..i0+15, K=0..127 (16-bit A 16x32 VGPR layout) ---
    union AF { v16h v; uint4 q[2]; };
    AF a[4];
    {
        const uint4* ap = (const uint4*)(fnh + (size_t)(i0 + laneLo) * DIM);
#pragma unroll
        for (int kb = 0; kb < 4; ++kb) {
            const int c0 = kb * 4 + laneHi;  // 8-half (16B) chunk
            a[kb].q[0] = ap[c0];             // K = kb*32 + laneHi*8 .. +7
            a[kb].q[1] = ap[c0 + 2];         // K = .. + 16
        }
    }

    int   labi[8];
    float Er[8];
#pragma unroll
    for (int r = 0; r < 8; ++r) {
        const int gi = i0 + r + laneHi * 8;
        labi[r] = labels[gi];
        Er[r]   = (pass == 2) ? E[gi] : 0.0f;
    }

    float accE[8];
#pragma unroll
    for (int r = 0; r < 8; ++r) accE[r] = 0.0f;
    float accL = 0.0f, accN = 0.0f;

    // Prologue: TDM-stage tile 0 into buffer 0.
    if (wave == 0) {
        tdm_load_tile(fnh + (size_t)jBase * DIM, ldsOff0);
        __builtin_amdgcn_s_wait_tensorcnt(0);
    }
    __syncthreads();

    for (int t = 0; t < nTiles; ++t) {
        const int buf = t & 1;

        // Async-stage tile t+1 into the back buffer while we compute tile t.
        if (wave == 0 && t + 1 < nTiles)
            tdm_load_tile(fnh + (size_t)(jBase + (t + 1) * 16) * DIM,
                          buf ? ldsOff0 : ldsOff1);
        // Warm GL2 for tile t+2 (32 lanes x 128B covers the 4KB tile).
        if (wave == 0 && t + 2 < nTiles)
            __builtin_prefetch((const void*)(fnh + (size_t)(jBase + (t + 2) * 16) * DIM + lane * 64), 0, 3);

        // --- preload all 4 B fragments, then chain 4 WMMAs over K=128 ---
        const _Float16* bb = &ldsB[buf][laneLo * LDS_H + laneHi * 8];
        union BF { v16h v; uint4 q[2]; } b[4];
#pragma unroll
        for (int kb = 0; kb < 4; ++kb) {
            b[kb].q[0] = *(const uint4*)(bb + kb * 32);
            b[kb].q[1] = *(const uint4*)(bb + kb * 32 + 16);
        }
        v8f c = {};
#pragma unroll
        for (int kb = 0; kb < 4; ++kb)
            c = __builtin_amdgcn_wmma_f32_16x16x32_f16(
                    false, a[kb].v, false, b[kb].v, (short)0, c, false, false);

        const int labj = labels[jBase + t * 16 + laneLo];
        if (pass == 1) {
#pragma unroll
            for (int r = 0; r < 8; ++r) {
                const float e = __builtin_amdgcn_exp2f(c[r] * K_EXP2);   // exp(2c)
                accE[r] += (labi[r] == labj) ? 1.0f : e;
            }
        } else {
#pragma unroll
            for (int r = 0; r < 8; ++r) {
                const float sim = c[r] * 2.0f;
                const float e   = __builtin_amdgcn_exp2f(c[r] * K_EXP2); // exp(sim)
                const bool pos  = (labi[r] == labj);
                const float lv  = __builtin_amdgcn_logf(e + Er[r]) * LN2 - sim;
                accL += pos ? lv : 0.0f;
                accN += pos ? 1.0f : 0.0f;
            }
        }

        // Issuing wave drains its TENSORcnt, then barrier releases the block.
        if (wave == 0 && t + 1 < nTiles)
            __builtin_amdgcn_s_wait_tensorcnt(0);
        __syncthreads();
    }

    if (pass == 1) {
#pragma unroll
        for (int r = 0; r < 8; ++r) {
            float e = accE[r];
#pragma unroll
            for (int m = 8; m >= 1; m >>= 1) e += __shfl_xor(e, m, 32);
            if (laneLo == 0)
                Epart[(size_t)blockIdx.y * N_ROWS + i0 + r + laneHi * 8] = e;
        }
    } else {
#pragma unroll
        for (int m = 16; m >= 1; m >>= 1) {
            accL += __shfl_xor(accL, m, 32);
            accN += __shfl_xor(accN, m, 32);
        }
        if (lane == 0) { redL[wave] = accL; redN[wave] = accN; }
        __syncthreads();
        if (threadIdx.x == 0) {
            float L = 0.0f, Np = 0.0f;
#pragma unroll
            for (int w = 0; w < 8; ++w) { L += redL[w]; Np += redN[w]; }
            const int bid = blockIdx.y * gridDim.x + blockIdx.x;
            lossPart[bid] = L;
            npPart[bid]   = Np;
        }
    }
}

// ---------------------------------------------------------------------------
// Kernel 3: E[i] = sum over JSPLIT column-chunk partials (fixed order).
// ---------------------------------------------------------------------------
__global__ __launch_bounds__(256)
void conloss_reduceE_kernel(const float* __restrict__ Epart, float* __restrict__ E) {
    const int i = blockIdx.x * 256 + threadIdx.x;
    float s = 0.0f;
#pragma unroll
    for (int k = 0; k < JSPLIT; ++k) s += Epart[(size_t)k * N_ROWS + i];
    E[i] = s;
}

// ---------------------------------------------------------------------------
// Kernel 4: final deterministic reduction -> scalar loss.
// ---------------------------------------------------------------------------
__global__ __launch_bounds__(256)
void conloss_final_kernel(const float* __restrict__ lossPart,
                          const float* __restrict__ npPart,
                          float* __restrict__ out) {
    __shared__ float sL[256], sN[256];
    const int t = threadIdx.x;
    float l = 0.0f, n = 0.0f;
#pragma unroll
    for (int k = t; k < NPART; k += 256) { l += lossPart[k]; n += npPart[k]; }
    sL[t] = l; sN[t] = n;
    __syncthreads();
    for (int s = 128; s > 0; s >>= 1) {
        if (t < s) { sL[t] += sL[t + s]; sN[t] += sN[t + s]; }
        __syncthreads();
    }
    if (t == 0) out[0] = sL[0] / sN[0];
}

// ---------------------------------------------------------------------------
extern "C" void kernel_launch(void* const* d_in, const int* in_sizes, int n_in,
                              void* d_out, int out_size, void* d_ws, size_t ws_size,
                              hipStream_t stream) {
    (void)in_sizes; (void)n_in; (void)out_size; (void)ws_size;
    const float* F      = (const float*)d_in[0];
    const int*   labels = (const int*)d_in[1];
    float*       out    = (float*)d_out;

    char* ws = (char*)d_ws;
    _Float16* fnh = (_Float16*)ws;
    size_t off = (size_t)N_ROWS * DIM * sizeof(_Float16);          // 2 MB
    float* Epart = (float*)(ws + off); off += (size_t)JSPLIT * N_ROWS * sizeof(float);
    float* E     = (float*)(ws + off); off += (size_t)N_ROWS * sizeof(float);
    float* lossPart = (float*)(ws + off); off += (size_t)NPART * sizeof(float);
    float* npPart   = (float*)(ws + off);

    conloss_norm_kernel<<<N_ROWS / 8, 256, 0, stream>>>(F, fnh);

    dim3 grid(N_ROWS / 128, JSPLIT);
    conloss_sim_kernel<<<grid, 256, 0, stream>>>(fnh, labels, Epart, E, lossPart, npPart, 1);
    conloss_reduceE_kernel<<<N_ROWS / 256, 256, 0, stream>>>(Epart, E);
    conloss_sim_kernel<<<grid, 256, 0, stream>>>(fnh, labels, Epart, E, lossPart, npPart, 2);
    conloss_final_kernel<<<1, 256, 0, stream>>>(lossPart, npPart, out);
}
